// AtomLevel_7275674599784
// MI455X (gfx1250) — matrix operand order repaired and checked by reading, not compiled, not run on hardware
//
#include <hip/hip_runtime.h>
#include <hip/hip_bf16.h>
#include <math.h>

#define NN 16384
#define EE 262144
#define HH 128
#define BB 32
#define RR 50
#define LL 4

typedef __bf16 bf16;
typedef __attribute__((ext_vector_type(16))) __bf16 v16bf;
typedef __attribute__((ext_vector_type(8)))  __bf16 v8bf;
typedef __attribute__((ext_vector_type(8)))  float  v8f;
typedef __attribute__((ext_vector_type(4)))  unsigned u32x4;
typedef __attribute__((ext_vector_type(8)))  int      i32x8;
typedef __attribute__((ext_vector_type(4)))  int      i32x4;

#if __has_builtin(__builtin_amdgcn_tensor_load_to_lds) && \
    __has_builtin(__builtin_amdgcn_s_wait_tensorcnt)
#define USE_TDM 1
#endif

// ---------------------------------------------------------------------------
// A-operand fetch for the generic 128-col GEMM.
//  AMODE 1: plain bf16 matrix A[row*lda + k]
//  AMODE 2: edge concat  [ x[dst] | x[src] | ee ]   (K = 384)
//  AMODE 3: node concat  [ aggr(f32) | x(bf16) ]    (K = 256)
// ---------------------------------------------------------------------------
template<int AMODE>
__device__ __forceinline__ float fetchA(int row, int k,
    const bf16* Ab, int lda,
    const bf16* Xb, const bf16* Eb, const int* idx0, const int* idx1,
    const float* Aggr)
{
  if constexpr (AMODE == 1) {
    return (float)Ab[(size_t)row * lda + k];
  } else if constexpr (AMODE == 2) {
    if (k < HH)          return (float)Xb[(size_t)idx0[row] * HH + k];
    else if (k < 2 * HH) return (float)Xb[(size_t)idx1[row] * HH + (k - HH)];
    else                 return (float)Eb[(size_t)row * HH + (k - 2 * HH)];
  } else { // AMODE == 3
    if (k < HH) return Aggr[(size_t)row * HH + k];
    else        return (float)Xb[(size_t)row * HH + (k - HH)];
  }
}

// ---------------------------------------------------------------------------
// Generic tiled GEMM:  C[M,128] = act(A[M,K] @ W[K,128] + bias)
// Weights are pre-transposed+converted: Wt[128][K] bf16 (row = output col).
// Block = 256 threads (8 waves), tile = 128x128, K-step = 32, bf16 WMMA.
// B tile [128 cols x 32 k] is DMA'd into LDS by the Tensor Data Mover
// (wave 0 issues tensor_load_to_lds; tile_dim0=40 with tensor_dim0=K-k0 gives
// zero-filled 40-element pitch rows -> identical layout to the manual fill).
// ACT: 0 none, 1 silu, 2 sigmoid, 3 tanh.  OUTF32: write fp32 else bf16.
// ---------------------------------------------------------------------------
template<int AMODE, int ACT, int OUTF32>
__global__ __launch_bounds__(256)
void gemm128_kernel(const bf16* Ab, int lda, int K,
                    const bf16* Xb, const bf16* Eb,
                    const int* idx0, const int* idx1,
                    const float* Aggr,
                    const bf16* Wt, const float* bias,
                    float* Cf, bf16* Cb)
{
  __shared__ __attribute__((aligned(16))) bf16 sA[128 * 40];
  __shared__ __attribute__((aligned(16))) bf16 sB[128 * 40];

  const int tid  = threadIdx.x;
  const int w    = tid >> 5;        // wave id 0..7 -> output row strip
  const int lane = tid & 31;
  const int hh   = lane >> 4;       // half-lane
  const int m    = lane & 15;
  const int row0 = blockIdx.x * 128;

  v8f acc[8] = {};

  const int nk = (K + 31) >> 5;
  for (int kk = 0; kk < nk; ++kk) {
    const int k0 = kk << 5;
    __syncthreads();  // previous iteration's LDS reads are complete

#ifdef USE_TDM
    if (tid < 32) {   // wave 0 only: issue one TDM transfer for the B tile
      const unsigned ldsb = (unsigned)(size_t)&sB[0];
      const unsigned long long ga =
          (unsigned long long)(size_t)(Wt + k0);        // tile start (bytes)
      const unsigned td0 = (unsigned)(K - k0);          // right-edge zero fill
      u32x4 g0;
      g0[0] = 1u;                                       // count=1, user mode
      g0[1] = ldsb;                                     // lds_addr
      g0[2] = (unsigned)ga;                             // global_addr[31:0]
      g0[3] = (unsigned)((ga >> 32) & 0x01FFFFFFull) | (2u << 30); // type=2
      i32x8 g1;
      g1[0] = 0x00010000;                               // data_size=1 (2 B)
      g1[1] = (int)((td0 & 0xFFFFu) << 16);             // tensor_dim0 lo
      g1[2] = (int)((td0 >> 16) | (128u << 16));        // dim0 hi | tensor_dim1
      g1[3] = (int)(40u << 16);                         // tile_dim0 = 40
      g1[4] = 128;                                      // tile_dim1 = 128
      g1[5] = K;                                        // tensor_dim0_stride lo
      g1[6] = 0;
      g1[7] = 0;
      const i32x4 z4 = {0, 0, 0, 0};
#if __has_include(<hip/amd_detail/amd_gfx1250_TDM.h>)
      const i32x8 z8 = {0, 0, 0, 0, 0, 0, 0, 0};
      __builtin_amdgcn_tensor_load_to_lds(g0, g1, z4, z4, z8, 0);
#else
      __builtin_amdgcn_tensor_load_to_lds(g0, g1, z4, z4, 0);
#endif
    }
#endif

    // Stage A tile (128 rows x 32 k) as bf16, row pitch 40 (all waves).
    for (int i = tid; i < 128 * 32; i += 256) {
      const int r = i >> 5, k = i & 31;
      const int gk = k0 + k;
      float v = 0.0f;
      if (gk < K) v = fetchA<AMODE>(row0 + r, gk, Ab, lda, Xb, Eb, idx0, idx1, Aggr);
      sA[r * 40 + k] = (bf16)v;
    }

#ifdef USE_TDM
    if (tid < 32) __builtin_amdgcn_s_wait_tensorcnt(0);
#else
    // Fallback: manual B staging, sB[col][k], col pitch 40.
    for (int i = tid; i < 32 * 128; i += 256) {
      const int k = i >> 7, n = i & 127;
      const int gk = k0 + k;
      sB[n * 40 + k] = (gk < K) ? Wt[(size_t)n * K + gk] : (bf16)0.0f;
    }
#endif
    if (kk + 1 < nk) __builtin_prefetch(&Wt[(size_t)(k0 + 32)], 0, 1);
    __syncthreads();

    // A fragment: row = 16w + m; ISA layout K(e,h) = (e&7) + 8h + 16*(e>>3)
    // -> bytes [row*80 + 16h, 16) and [row*80 + 32 + 16h, 16)
    const v8bf* pa = (const v8bf*)(sA + (16 * w + m) * 40 + 8 * hh);
    const v16bf afrag = __builtin_shufflevector(pa[0], pa[2],
        0,1,2,3,4,5,6,7,8,9,10,11,12,13,14,15);

    // Load all 8 B fragments first, then stream 8 WMMAs back-to-back.
    v16bf bfrag[8];
#pragma unroll
    for (int j = 0; j < 8; ++j) {
      // B fragment: col = 16j + m; K(e,h) = (e&7) + 8h + 16*(e>>3)
      const v8bf* pb = (const v8bf*)(sB + (16 * j + m) * 40 + 8 * hh);
      bfrag[j] = __builtin_shufflevector(pb[0], pb[2],
          0,1,2,3,4,5,6,7,8,9,10,11,12,13,14,15);
    }
#pragma unroll
    for (int j = 0; j < 8; ++j) {
      acc[j] = __builtin_amdgcn_wmma_f32_16x16x32_bf16(
          false, afrag, false, bfrag[j], (short)0, acc[j], false, false);
    }
  }

  // Epilogue: bias + activation + store.  C/D layout: M = r + 8h, N = m.
#pragma unroll
  for (int j = 0; j < 8; ++j) {
    const int gc = 16 * j + m;
    const float bi = bias ? bias[gc] : 0.0f;
#pragma unroll
    for (int r = 0; r < 8; ++r) {
      float v = acc[j][r] + bi;
      if constexpr (ACT == 1)      v = v / (1.0f + __expf(-v));
      else if constexpr (ACT == 2) v = 1.0f / (1.0f + __expf(-v));
      else if constexpr (ACT == 3) v = tanhf(v);
      const int gr = row0 + 16 * w + 8 * hh + r;
      if constexpr (OUTF32) Cf[(size_t)gr * 128 + gc] = v;
      else                  Cb[(size_t)gr * 128 + gc] = (bf16)v;
    }
  }
}

// ---------------------------------------------------------------------------
// Weight prep: W[l][K][128] f32  ->  Wt[l][128][K] bf16 (transposed rows)
// ---------------------------------------------------------------------------
__global__ void wtrans_kernel(const float* W, bf16* Wt, int K, int layers)
{
  const long total = (long)layers * K * 128;
  const long i = (long)blockIdx.x * blockDim.x + threadIdx.x;
  if (i >= total) return;
  const int  l = (int)(i / ((long)K * 128));
  const int  r = (int)(i % ((long)K * 128));
  const int  n = r / K, k = r % K;
  Wt[i] = (bf16)W[(size_t)l * K * 128 + (size_t)k * 128 + n];
}

// ---------------------------------------------------------------------------
// Small elementwise / scatter / reduce kernels
// ---------------------------------------------------------------------------
__global__ void embed_kernel(const int* an, const float* embed, float* xf, bf16* xb)
{
  const int i = blockIdx.x * blockDim.x + threadIdx.x;
  if (i >= NN * HH) return;
  const int n = i >> 7, h = i & 127;
  const float v = embed[(size_t)an[n] * HH + h];
  xf[i] = v; xb[i] = (bf16)v;
}

__global__ void zero_kernel(float* p, int n)
{
  const int i = blockIdx.x * blockDim.x + threadIdx.x;
  if (i < n) p[i] = 0.0f;
}

__global__ void rbf_kernel(const float* pos, const int* src, const int* dst,
                           const float* widths, bf16* ea)
{
  const int e = blockIdx.x * blockDim.x + threadIdx.x;
  if (e >= EE) return;
  const int s = src[e], d2 = dst[e];
  const float dx = pos[d2 * 3 + 0] - pos[s * 3 + 0];
  const float dy = pos[d2 * 3 + 1] - pos[s * 3 + 1];
  const float dz = pos[d2 * 3 + 2] - pos[s * 3 + 2];
  const float d = sqrtf(dx * dx + dy * dy + dz * dz);
  const float cut = (d < 5.0f)
      ? 0.5f * (__cosf(d * (3.14159265358979f / 5.0f)) + 1.0f) : 0.0f;
#pragma unroll 5
  for (int r = 0; r < RR; ++r) {
    const float c = (float)r * (5.0f / 49.0f);
    const float wv = widths[r];
    const float gamma = 1.0f / (2.0f * wv * wv);
    const float t = d - c;
    ea[(size_t)e * RR + r] = (bf16)(__expf(-gamma * t * t) * cut);
  }
}

__global__ void scatter_kernel(const bf16* msg, const bf16* gate,
                               const int* dst, float* aggr)
{
  const int i = blockIdx.x * blockDim.x + threadIdx.x;
  if (i >= EE * HH) return;
  const int e = i >> 7, h = i & 127;
  const float v = (float)msg[i] * (float)gate[i];
  atomicAdd(&aggr[(size_t)dst[e] * HH + h], v);
}

__global__ void lnres_kernel(float* x, const float* upd, const float* g,
                             const float* b, bf16* xb, float* fsum, int accum)
{
  __shared__ float red[128];
  const int n = blockIdx.x, t = threadIdx.x;
  const size_t i = (size_t)n * HH + t;
  const float u = x[i] + upd[i];
  red[t] = u; __syncthreads();
  for (int s = 64; s > 0; s >>= 1) { if (t < s) red[t] += red[t + s]; __syncthreads(); }
  const float mean = red[0] * (1.0f / 128.0f);
  __syncthreads();
  const float dv = u - mean;
  red[t] = dv * dv; __syncthreads();
  for (int s = 64; s > 0; s >>= 1) { if (t < s) red[t] += red[t + s]; __syncthreads(); }
  const float var = red[0] * (1.0f / 128.0f);
  const float y = dv * rsqrtf(var + 1e-5f) * g[t] + b[t];
  x[i] = y; xb[i] = (bf16)y;
  if (accum) fsum[i] += y;
}

__global__ void atomfeat_kernel(const float* fsum, float* af, bf16* afb)
{
  const int i = blockIdx.x * blockDim.x + threadIdx.x;
  if (i >= NN * HH) return;
  const float v = fsum[i] * (1.0f / 3.0f);
  af[i] = v; afb[i] = (bf16)v;
}

__global__ void cvt_kernel(const float* s, bf16* d, int n)
{
  const int i = blockIdx.x * blockDim.x + threadIdx.x;
  if (i < n) d[i] = (bf16)s[i];
}

__global__ void gdot_kernel(const bf16* gh, const float* w2, const float* b2, float* g)
{
  const int n = blockIdx.x * blockDim.x + threadIdx.x;
  if (n >= NN) return;
  float s = b2[0];
#pragma unroll 8
  for (int k = 0; k < HH; ++k) s += (float)gh[(size_t)n * HH + k] * w2[k];
  g[n] = s;
}

__device__ __forceinline__ unsigned enc_f32(float f)
{
  const unsigned u = __float_as_uint(f);
  return (u & 0x80000000u) ? ~u : (u | 0x80000000u);
}
__device__ __forceinline__ float dec_f32(unsigned e)
{
  return (e & 0x80000000u) ? __uint_as_float(e ^ 0x80000000u)
                           : __uint_as_float(~e);
}

__global__ void gmax_init_kernel(unsigned* gm)
{
  const int b = threadIdx.x;
  if (b < BB) gm[b] = enc_f32(-INFINITY);
}

__global__ void gmax_atomic_kernel(const float* g, const int* batch, unsigned* gm)
{
  const int n = blockIdx.x * blockDim.x + threadIdx.x;
  if (n >= NN) return;
  atomicMax(&gm[batch[n]], enc_f32(g[n]));
}

__global__ void gmax_decode_kernel(const unsigned* gm, float* gmaxf, float* den)
{
  const int b = threadIdx.x;
  if (b >= BB) return;
  float v = dec_f32(gm[b]);
  if (!isfinite(v)) v = 0.0f;
  gmaxf[b] = v;
  den[b] = 0.0f;
}

__global__ void exden_kernel(const float* g, const int* batch,
                             const float* gmaxf, float* ex, float* den)
{
  const int n = blockIdx.x * blockDim.x + threadIdx.x;
  if (n >= NN) return;
  const float e = __expf(g[n] - gmaxf[batch[n]]);
  ex[n] = e;
  atomicAdd(&den[batch[n]], e);
}

__global__ void final_kernel(const bf16* hb, const float* ex, const float* den,
                             const int* batch, float* out)
{
  const int i = blockIdx.x * blockDim.x + threadIdx.x;
  if (i >= NN * HH) return;
  const int n = i >> 7, h = i & 127;
  const float a = ex[n] / den[batch[n]];
  atomicAdd(&out[(size_t)batch[n] * HH + h], a * (float)hb[i]);
}

// ---------------------------------------------------------------------------
// Host orchestration
// ---------------------------------------------------------------------------
extern "C" void kernel_launch(void* const* d_in, const int* in_sizes, int n_in,
                              void* d_out, int out_size, void* d_ws, size_t ws_size,
                              hipStream_t stream)
{
  (void)in_sizes; (void)n_in; (void)out_size; (void)ws_size;

  const int*   an     = (const int*)d_in[0];
  const float* pos    = (const float*)d_in[1];
  const int*   batch  = (const int*)d_in[2];
  const int*   eidx   = (const int*)d_in[3];
  const float* embed  = (const float*)d_in[4];
  const float* widths = (const float*)d_in[5];
  const float* ee_w1  = (const float*)d_in[6];
  const float* ee_b1  = (const float*)d_in[7];
  const float* ee_w2  = (const float*)d_in[8];
  const float* ee_b2  = (const float*)d_in[9];
  const float* msg_w1 = (const float*)d_in[10];
  const float* msg_b1 = (const float*)d_in[11];
  const float* msg_w2 = (const float*)d_in[12];
  const float* msg_b2 = (const float*)d_in[13];
  const float* gate_w = (const float*)d_in[14];
  const float* gate_b = (const float*)d_in[15];
  const float* upd_w1 = (const float*)d_in[16];
  const float* upd_b1 = (const float*)d_in[17];
  const float* upd_w2 = (const float*)d_in[18];
  const float* upd_b2 = (const float*)d_in[19];
  const float* ln_g   = (const float*)d_in[20];
  const float* ln_b   = (const float*)d_in[21];
  const float* gp_w1  = (const float*)d_in[22];
  const float* gp_b1  = (const float*)d_in[23];
  const float* gp_w2  = (const float*)d_in[24];
  const float* gp_b2  = (const float*)d_in[25];
  const float* pn_w   = (const float*)d_in[26];
  const float* pn_b   = (const float*)d_in[27];
  const float* ro_w1  = (const float*)d_in[28];
  const float* ro_b1  = (const float*)d_in[29];
  const float* ro_w2  = (const float*)d_in[30];
  const float* ro_b2  = (const float*)d_in[31];

  const int* src = eidx;        // edge_index[0]
  const int* dst = eidx + EE;   // edge_index[1]

  // Workspace carve-up (256B aligned).
  char* ws = (char*)d_ws;
  size_t off = 0;
  auto alloc = [&](size_t bytes) -> char* {
    char* p = ws + off;
    off = (off + bytes + 255) & ~(size_t)255;
    return p;
  };
  bf16*  eab  = (bf16*)alloc((size_t)EE * RR * sizeof(bf16));   // edge_attr
  bf16*  B1   = (bf16*)alloc((size_t)EE * HH * sizeof(bf16));   // eh / mh / gate
  bf16*  B2   = (bf16*)alloc((size_t)EE * HH * sizeof(bf16));   // ee
  bf16*  B3   = (bf16*)alloc((size_t)EE * HH * sizeof(bf16));   // msg
  float* xf   = (float*)alloc((size_t)NN * HH * sizeof(float));
  bf16*  xb   = (bf16*)alloc((size_t)NN * HH * sizeof(bf16));
  float* aggr = (float*)alloc((size_t)NN * HH * sizeof(float));
  bf16*  nb1  = (bf16*)alloc((size_t)NN * HH * sizeof(bf16));
  float* updf = (float*)alloc((size_t)NN * HH * sizeof(float));
  float* fsum = (float*)alloc((size_t)NN * HH * sizeof(float));
  bf16*  afb  = (bf16*)alloc((size_t)NN * HH * sizeof(bf16));
  bf16*  aeb  = (bf16*)alloc((size_t)NN * HH * sizeof(bf16));
  bf16*  ghb  = (bf16*)alloc((size_t)NN * HH * sizeof(bf16));
  bf16*  hb   = (bf16*)alloc((size_t)NN * HH * sizeof(bf16));
  float* gbuf = (float*)alloc((size_t)NN * sizeof(float));
  float* exbuf= (float*)alloc((size_t)NN * sizeof(float));
  unsigned* gmaxu = (unsigned*)alloc((size_t)BB * sizeof(unsigned));
  float* gmaxf = (float*)alloc((size_t)BB * sizeof(float));
  float* den   = (float*)alloc((size_t)BB * sizeof(float));
  // Transposed bf16 weights for the TDM-fed GEMM B operand.
  bf16* wt_ee1  = (bf16*)alloc((size_t)LL * RR * HH * sizeof(bf16));
  bf16* wt_ee2  = (bf16*)alloc((size_t)LL * HH * HH * sizeof(bf16));
  bf16* wt_msg1 = (bf16*)alloc((size_t)LL * 3 * HH * HH * sizeof(bf16));
  bf16* wt_msg2 = (bf16*)alloc((size_t)LL * HH * HH * sizeof(bf16));
  bf16* wt_gate = (bf16*)alloc((size_t)LL * HH * HH * sizeof(bf16));
  bf16* wt_upd1 = (bf16*)alloc((size_t)LL * 2 * HH * HH * sizeof(bf16));
  bf16* wt_upd2 = (bf16*)alloc((size_t)LL * HH * HH * sizeof(bf16));
  bf16* wt_gp1  = (bf16*)alloc((size_t)HH * HH * sizeof(bf16));
  bf16* wt_pn   = (bf16*)alloc((size_t)HH * HH * sizeof(bf16));
  bf16* wt_ro1  = (bf16*)alloc((size_t)HH * HH * sizeof(bf16));
  bf16* wt_ro2  = (bf16*)alloc((size_t)HH * HH * sizeof(bf16));

  float* out_af = (float*)d_out;                       // [N,H]
  float* out_ae = out_af + (size_t)NN * HH;            // [N,H]
  float* out_gf = out_ae + (size_t)NN * HH;            // [B,H]

  const int T = 256;
  const dim3 gEdge(EE / 128), gNode(NN / 128);
  auto wgrid = [&](long K, long layers) {
    return dim3((unsigned)((layers * K * 128 + T - 1) / T));
  };

  // Weight prep (transpose + bf16 convert)
  wtrans_kernel<<<wgrid(RR, LL),     T, 0, stream>>>(ee_w1,  wt_ee1,  RR,     LL);
  wtrans_kernel<<<wgrid(HH, LL),     T, 0, stream>>>(ee_w2,  wt_ee2,  HH,     LL);
  wtrans_kernel<<<wgrid(3 * HH, LL), T, 0, stream>>>(msg_w1, wt_msg1, 3 * HH, LL);
  wtrans_kernel<<<wgrid(HH, LL),     T, 0, stream>>>(msg_w2, wt_msg2, HH,     LL);
  wtrans_kernel<<<wgrid(HH, LL),     T, 0, stream>>>(gate_w, wt_gate, HH,     LL);
  wtrans_kernel<<<wgrid(2 * HH, LL), T, 0, stream>>>(upd_w1, wt_upd1, 2 * HH, LL);
  wtrans_kernel<<<wgrid(HH, LL),     T, 0, stream>>>(upd_w2, wt_upd2, HH,     LL);
  wtrans_kernel<<<wgrid(HH, 1),      T, 0, stream>>>(gp_w1,  wt_gp1,  HH,     1);
  wtrans_kernel<<<wgrid(HH, 1),      T, 0, stream>>>(pn_w,   wt_pn,   HH,     1);
  wtrans_kernel<<<wgrid(HH, 1),      T, 0, stream>>>(ro_w1,  wt_ro1,  HH,     1);
  wtrans_kernel<<<wgrid(HH, 1),      T, 0, stream>>>(ro_w2,  wt_ro2,  HH,     1);

  embed_kernel<<<(NN * HH + T - 1) / T, T, 0, stream>>>(an, embed, xf, xb);
  zero_kernel<<<(NN * HH + T - 1) / T, T, 0, stream>>>(fsum, NN * HH);
  rbf_kernel<<<(EE + T - 1) / T, T, 0, stream>>>(pos, src, dst, widths, eab);

  for (int l = 0; l < LL; ++l) {
    // ee = MLP2(edge_attr)
    gemm128_kernel<1, 1, 0><<<gEdge, T, 0, stream>>>(
        eab, RR, RR, nullptr, nullptr, nullptr, nullptr, nullptr,
        wt_ee1 + (size_t)l * RR * HH, ee_b1 + (size_t)l * HH, nullptr, B1);
    gemm128_kernel<1, 0, 0><<<gEdge, T, 0, stream>>>(
        B1, HH, HH, nullptr, nullptr, nullptr, nullptr, nullptr,
        wt_ee2 + (size_t)l * HH * HH, ee_b2 + (size_t)l * HH, nullptr, B2);
    // msg = MLP2(concat(x[dst], x[src], ee))  — gather fused into A-loader
    gemm128_kernel<2, 1, 0><<<gEdge, T, 0, stream>>>(
        nullptr, 0, 3 * HH, xb, B2, dst, src, nullptr,
        wt_msg1 + (size_t)l * 3 * HH * HH, msg_b1 + (size_t)l * HH, nullptr, B1);
    gemm128_kernel<1, 0, 0><<<gEdge, T, 0, stream>>>(
        B1, HH, HH, nullptr, nullptr, nullptr, nullptr, nullptr,
        wt_msg2 + (size_t)l * HH * HH, msg_b2 + (size_t)l * HH, nullptr, B3);
    // gate = sigmoid(ee @ gate_w + b)
    gemm128_kernel<1, 2, 0><<<gEdge, T, 0, stream>>>(
        B2, HH, HH, nullptr, nullptr, nullptr, nullptr, nullptr,
        wt_gate + (size_t)l * HH * HH, gate_b + (size_t)l * HH, nullptr, B1);
    // aggr = segment_sum(msg * gate, dst)
    zero_kernel<<<(NN * HH + T - 1) / T, T, 0, stream>>>(aggr, NN * HH);
    scatter_kernel<<<(EE * HH + T - 1) / T, T, 0, stream>>>(B3, B1, dst, aggr);
    // upd = MLP2(concat(aggr, x)) — gather fused
    gemm128_kernel<3, 1, 0><<<gNode, T, 0, stream>>>(
        nullptr, 0, 2 * HH, xb, nullptr, nullptr, nullptr, aggr,
        wt_upd1 + (size_t)l * 2 * HH * HH, upd_b1 + (size_t)l * HH, nullptr, nb1);
    gemm128_kernel<1, 0, 1><<<gNode, T, 0, stream>>>(
        nb1, HH, HH, nullptr, nullptr, nullptr, nullptr, nullptr,
        wt_upd2 + (size_t)l * HH * HH, upd_b2 + (size_t)l * HH, updf, nullptr);
    // x = LN(x + upd); accumulate last-3 feature sum
    lnres_kernel<<<NN, 128, 0, stream>>>(
        xf, updf, ln_g + (size_t)l * HH, ln_b + (size_t)l * HH,
        xb, fsum, (l >= 1) ? 1 : 0);
  }

  // atom_features = mean of last 3 layer outputs
  atomfeat_kernel<<<(NN * HH + T - 1) / T, T, 0, stream>>>(fsum, out_af, afb);
  // atom_embeddings = MLP2(atom_features, ro)
  gemm128_kernel<1, 1, 0><<<gNode, T, 0, stream>>>(
      afb, HH, HH, nullptr, nullptr, nullptr, nullptr, nullptr,
      wt_ro1, ro_b1, nullptr, nb1);
  gemm128_kernel<1, 0, 1><<<gNode, T, 0, stream>>>(
      nb1, HH, HH, nullptr, nullptr, nullptr, nullptr, nullptr,
      wt_ro2, ro_b2, out_ae, nullptr);
  cvt_kernel<<<(NN * HH + T - 1) / T, T, 0, stream>>>(out_ae, aeb, NN * HH);
  // attention gate: g = (tanh(ae@gp_w1+b1)) @ gp_w2 + b2
  gemm128_kernel<1, 3, 0><<<gNode, T, 0, stream>>>(
      aeb, HH, HH, nullptr, nullptr, nullptr, nullptr, nullptr,
      wt_gp1, gp_b1, nullptr, ghb);
  gdot_kernel<<<(NN + T - 1) / T, T, 0, stream>>>(ghb, gp_w2, gp_b2, gbuf);
  gmax_init_kernel<<<1, BB, 0, stream>>>(gmaxu);
  gmax_atomic_kernel<<<(NN + T - 1) / T, T, 0, stream>>>(gbuf, batch, gmaxu);
  gmax_decode_kernel<<<1, BB, 0, stream>>>(gmaxu, gmaxf, den);
  exden_kernel<<<(NN + T - 1) / T, T, 0, stream>>>(gbuf, batch, gmaxf, exbuf, den);
  // h = silu(ae @ pn_w + pn_b); graph feats = segment_sum(alpha*h)
  gemm128_kernel<1, 1, 0><<<gNode, T, 0, stream>>>(
      aeb, HH, HH, nullptr, nullptr, nullptr, nullptr, nullptr,
      wt_pn, pn_b, nullptr, hb);
  zero_kernel<<<(BB * HH + T - 1) / T, T, 0, stream>>>(out_gf, BB * HH);
  final_kernel<<<(NN * HH + T - 1) / T, T, 0, stream>>>(hb, exbuf, den, batch, out_gf);
}
